// GPT2CausalSelfAttention_11484742550166
// MI455X (gfx1250) — matrix operand order repaired and checked by reading, not compiled, not run on hardware
//
#include <hip/hip_runtime.h>
#include <hip/hip_bf16.h>

// ---------------------------------------------------------------------------
// GPT-2 causal self-attention for gfx1250 (MI455X).
// bf16 WMMA everywhere (v_wmma_f32_16x16x32_bf16), fp32 softmax.
// GEMMs: 256M x 64N per block (8 waves, 32M x 64N per wave), shared B tile
// staged through double-buffered LDS with global_load_async_to_lds_b128
// (ASYNCcnt) + s_wait_asynccnt + workgroup barrier.
// ---------------------------------------------------------------------------

typedef __attribute__((ext_vector_type(16))) __bf16         v16bf;
typedef __attribute__((ext_vector_type(8)))  float          v8f;
typedef __attribute__((ext_vector_type(16))) unsigned short u16x16;
typedef __attribute__((ext_vector_type(8)))  unsigned short u16x8;

#define DEV __device__ __forceinline__

static constexpr int Bb = 4, Tt = 2048, Cc = 768, Hh = 12, Dh = 64;
static constexpr int BT = Bb * Tt;   // 8192 tokens
static constexpr int C3 = 3 * Cc;    // 2304
static constexpr int BSTR = 40;      // LDS B-tile row stride (ushorts), padded

// round-to-nearest-even fp32 -> bf16
DEV unsigned short f2bf(float f) {
  union { float f; unsigned u; } v; v.f = f;
  unsigned u = v.u;
  unsigned r = u + 0x7FFFu + ((u >> 16) & 1u);
  return (unsigned short)(r >> 16);
}

DEV v8f wmma_bf16(u16x16 a, u16x16 b, v8f c) {
  return __builtin_amdgcn_wmma_f32_16x16x32_bf16(
      false, __builtin_bit_cast(v16bf, a),
      false, __builtin_bit_cast(v16bf, b),
      (short)0, c, false, false);
}

// 16-bit A-matrix 16x32 frag: lane m=lane%16; lanes<16 hold
// K = 0..15, lanes>=16 hold K = 16..31 (matches B frag order).
DEV u16x16 load_a_frag(const unsigned short* rowp, int lane) {
  const unsigned short* p = rowp + ((lane & 16) ? 16 : 0);
  union { u16x16 v; u16x8 h[2]; } u;
  u.h[0] = *(const u16x8*)(p);
  u.h[1] = *(const u16x8*)(p + 8);
  return u.v;
}

// 16-bit B-matrix 32x16 frag from global [N,K] bf16: lane n=lane%16;
// lanes<16 hold K=0..15, lanes>=16 hold K=16..31 (contiguous 32B).
DEV u16x16 load_b_frag(const unsigned short* colp, int lane) {
  return *(const u16x16*)(colp + ((lane & 16) ? 16 : 0));
}

// Same B frag but from an LDS column of 32 K-values (16B-aligned halves).
DEV u16x16 load_b_frag_lds(const unsigned short* colp, int lane) {
  const unsigned short* p = colp + ((lane & 16) ? 16 : 0);
  union { u16x16 v; u16x8 h[2]; } u;
  u.h[0] = *(const u16x8*)(p);
  u.h[1] = *(const u16x8*)(p + 8);
  return u.v;
}

// CDNA5 async copy: 16 bytes global -> LDS, tracked by ASYNCcnt.
DEV void async_copy16(unsigned ldsOff, const unsigned short* gsrc) {
  unsigned long long ga = (unsigned long long)(size_t)gsrc;
  asm volatile("global_load_async_to_lds_b128 %0, %1, off"
               :: "v"(ldsOff), "v"(ga) : "memory");
}
DEV void wait_async0() {
  asm volatile("s_wait_asynccnt 0" ::: "memory");
}

// ------------------------------- conversions -------------------------------

__global__ __launch_bounds__(256) void cvt_f32_bf16(const float* __restrict__ src,
                                                    unsigned short* __restrict__ dst,
                                                    int n) {
  for (int i = blockIdx.x * blockDim.x + threadIdx.x; i < n;
       i += gridDim.x * blockDim.x)
    dst[i] = f2bf(src[i]);
}

// w: [K,N] fp32 row-major -> wT: [N,K] bf16 (K contiguous = B-frag layout)
__global__ __launch_bounds__(256) void transpose_f32_bf16(const float* __restrict__ w,
                                                          unsigned short* __restrict__ wT,
                                                          int K, int N) {
  int total = K * N;
  for (int i = blockIdx.x * blockDim.x + threadIdx.x; i < total;
       i += gridDim.x * blockDim.x) {
    int n = i / K, k = i - n * K;
    wT[i] = f2bf(w[k * N + n]);
  }
}

// ------------------------------- QKV GEMM ----------------------------------
// [8192 x 768] @ [768 x 2304] + bias.  Block: 256M x 64N (8 waves x 32M).
// B tile (64 cols x 32 K) async-staged to double-buffered LDS.
// Q,K -> [B,H,T,Dh] bf16 (Dh contiguous);  V -> [B,H,Dh,T] bf16.

__global__ __launch_bounds__(256) void qkv_gemm(const unsigned short* __restrict__ xb,
                                                const unsigned short* __restrict__ wT,
                                                const float* __restrict__ bias,
                                                unsigned short* __restrict__ Qo,
                                                unsigned short* __restrict__ Ko,
                                                unsigned short* __restrict__ Vo) {
  __shared__ alignas(32) unsigned short Bsh[2][64 * BSTR];

  const int tid  = threadIdx.x;
  const int lane = tid & 31;
  const int wv   = tid >> 5;                // wave 0..7
  const int mBlk = blockIdx.x & 31;         // 32 M-blocks of 256 rows
  const int nBlk = blockIdx.x >> 5;         // 36 N-blocks of 64 cols
  const int lr = lane & 15, hi = lane >> 4;

  const unsigned short* wTn   = wT + (nBlk * 64) * Cc;
  const unsigned short* arow0 = xb + (mBlk * 256 + wv * 32 + lr) * Cc;
  const unsigned short* arow1 = arow0 + 16 * Cc;

  // this thread's 16-byte staging slice: column scol, segment sseg
  const int scol = tid >> 2, sseg = tid & 3;
  const unsigned short* sgsrc = wTn + scol * Cc + sseg * 8;
  const unsigned lb0 = (unsigned)(size_t)(&Bsh[0][scol * BSTR + sseg * 8]);
  const unsigned lb1 = (unsigned)(size_t)(&Bsh[1][scol * BSTR + sseg * 8]);

  v8f acc[8];
#pragma unroll
  for (int j = 0; j < 8; j++)
#pragma unroll
    for (int i = 0; i < 8; i++) acc[j][i] = 0.f;

  async_copy16(lb0, sgsrc);                 // prefetch k0 = 0

  for (int k0 = 0; k0 < Cc; k0 += 32) {
    const int p = (k0 >> 5) & 1;
    wait_async0();
    __syncthreads();
    if (k0 + 32 < Cc)
      async_copy16(p ? lb0 : lb1, sgsrc + k0 + 32);

    u16x16 a0 = load_a_frag(arow0 + k0, lane);
    u16x16 a1 = load_a_frag(arow1 + k0, lane);
    const unsigned short* bbuf = &Bsh[p][0];
#pragma unroll
    for (int j = 0; j < 4; j++) {
      u16x16 b = load_b_frag_lds(bbuf + (j * 16 + lr) * BSTR, lane);
      acc[j]     = wmma_bf16(a0, b, acc[j]);
      acc[4 + j] = wmma_bf16(a1, b, acc[4 + j]);
    }
  }

#pragma unroll
  for (int sub = 0; sub < 2; sub++) {
#pragma unroll
    for (int j = 0; j < 4; j++) {
      const int n    = nBlk * 64 + j * 16 + lr;
      const float bv = bias[n];
      const int sect = n / Cc;              // 0=Q 1=K 2=V
      const int cn   = n % Cc;
      const int h    = cn >> 6, d = cn & 63;
#pragma unroll
      for (int r = 0; r < 8; r++) {
        const int m    = mBlk * 256 + wv * 32 + sub * 16 + r + 8 * hi;
        const int bidx = m >> 11, t = m & (Tt - 1);
        const int bh   = bidx * Hh + h;
        const unsigned short val = f2bf(acc[sub * 4 + j][r] + bv);
        if (sect == 0)      Qo[(bh * Tt + t) * Dh + d] = val;
        else if (sect == 1) Ko[(bh * Tt + t) * Dh + d] = val;
        else                Vo[(bh * Dh + d) * Tt + t] = val;
      }
    }
  }
}

// ---------------------------- causal attention -----------------------------
// One wave per (b,h, 16-query tile).  Online softmax in C-layout row mapping;
// P re-laid C->A through per-wave LDS; 8 WMMAs per 32-key step.

__global__ __launch_bounds__(128) void attn_fwd(const unsigned short* __restrict__ Qo,
                                                const unsigned short* __restrict__ Ko,
                                                const unsigned short* __restrict__ Vo,
                                                unsigned short* __restrict__ Yo) {
  __shared__ unsigned short Pls[4][16 * 32];

  const int lane  = threadIdx.x & 31;
  const int wslot = threadIdx.x >> 5;
  const int wid   = blockIdx.x * 4 + wslot;
  const int qt    = wid & 127;
  const int bh    = wid >> 7;                 // 0..47
  const int b     = bh / Hh, h = bh % Hh;

  const unsigned short* qp = Qo + (size_t)bh * Tt * Dh;
  const unsigned short* kp = Ko + (size_t)bh * Tt * Dh;
  const unsigned short* vp = Vo + (size_t)bh * Dh * Tt;
  unsigned short* P = Pls[wslot];

  const int lr = lane & 15;
  const int hi = lane >> 4;
  const int qrow = qt * 16 + lr;

  const u16x16 qa0 = load_a_frag(qp + qrow * Dh, lane);
  const u16x16 qa1 = load_a_frag(qp + qrow * Dh + 32, lane);

  float mrow[8], lsum[8];
  v8f o0, o1, o2, o3;
#pragma unroll
  for (int r = 0; r < 8; r++) {
    mrow[r] = -__builtin_inff(); lsum[r] = 0.f;
    o0[r] = 0.f; o1[r] = 0.f; o2[r] = 0.f; o3[r] = 0.f;
  }

  const float NEGINF = -__builtin_inff();
  const float sc = 0.125f;                  // 1/sqrt(64)
  const int kend = (qt + 1) * 16;           // causal key bound

  for (int kb = 0; kb < kend; kb += 32) {
    v8f s0, s1;
#pragma unroll
    for (int i = 0; i < 8; i++) { s0[i] = 0.f; s1[i] = 0.f; }

    const unsigned short* kcol0 = kp + (kb + lr) * Dh;
    const unsigned short* kcol1 = kp + (kb + 16 + lr) * Dh;
    s0 = wmma_bf16(qa0, load_b_frag(kcol0, lane), s0);
    s0 = wmma_bf16(qa1, load_b_frag(kcol0 + 32, lane), s0);
    s1 = wmma_bf16(qa0, load_b_frag(kcol1, lane), s1);
    s1 = wmma_bf16(qa1, load_b_frag(kcol1 + 32, lane), s1);

#pragma unroll
    for (int r = 0; r < 8; r++) {
      const int qr = qt * 16 + r + 8 * hi;
      const int c0 = kb + lr, c1 = kb + 16 + lr;
      s0[r] = (c0 <= qr) ? s0[r] * sc : NEGINF;
      s1[r] = (c1 <= qr) ? s1[r] * sc : NEGINF;

      float tmax = fmaxf(s0[r], s1[r]);
      tmax = fmaxf(tmax, __shfl_xor(tmax, 1));
      tmax = fmaxf(tmax, __shfl_xor(tmax, 2));
      tmax = fmaxf(tmax, __shfl_xor(tmax, 4));
      tmax = fmaxf(tmax, __shfl_xor(tmax, 8));

      const float nm    = fmaxf(mrow[r], tmax);
      const float alpha = __expf(mrow[r] - nm);
      mrow[r] = nm;
      const float p0 = __expf(s0[r] - nm);
      const float p1 = __expf(s1[r] - nm);
      float rs = p0 + p1;
      rs += __shfl_xor(rs, 1);
      rs += __shfl_xor(rs, 2);
      rs += __shfl_xor(rs, 4);
      rs += __shfl_xor(rs, 8);
      lsum[r] = lsum[r] * alpha + rs;
      o0[r] *= alpha; o1[r] *= alpha; o2[r] *= alpha; o3[r] *= alpha;

      P[(r + 8 * hi) * 32 + lr]      = f2bf(p0);
      P[(r + 8 * hi) * 32 + 16 + lr] = f2bf(p1);
    }

    const u16x16 pa = load_a_frag(P + lr * 32, lane);
    o0 = wmma_bf16(pa, load_b_frag(vp + (0 * 16 + lr) * Tt + kb, lane), o0);
    o1 = wmma_bf16(pa, load_b_frag(vp + (1 * 16 + lr) * Tt + kb, lane), o1);
    o2 = wmma_bf16(pa, load_b_frag(vp + (2 * 16 + lr) * Tt + kb, lane), o2);
    o3 = wmma_bf16(pa, load_b_frag(vp + (3 * 16 + lr) * Tt + kb, lane), o3);
  }

#pragma unroll
  for (int r = 0; r < 8; r++) {
    const int t     = qt * 16 + r + 8 * hi;
    const float inv = 1.0f / lsum[r];
    const int base  = (b * Tt + t) * Cc + h * Dh;
    Yo[base + 0 * 16 + lr] = f2bf(o0[r] * inv);
    Yo[base + 1 * 16 + lr] = f2bf(o1[r] * inv);
    Yo[base + 2 * 16 + lr] = f2bf(o2[r] * inv);
    Yo[base + 3 * 16 + lr] = f2bf(o3[r] * inv);
  }
}

// ------------------------------ output proj --------------------------------
// [8192 x 768] @ [768 x 768] + bias -> fp32.  Same blocking as qkv_gemm.

__global__ __launch_bounds__(256) void proj_gemm(const unsigned short* __restrict__ yb,
                                                 const unsigned short* __restrict__ wT,
                                                 const float* __restrict__ bias,
                                                 float* __restrict__ out) {
  __shared__ alignas(32) unsigned short Bsh[2][64 * BSTR];

  const int tid  = threadIdx.x;
  const int lane = tid & 31;
  const int wv   = tid >> 5;
  const int mBlk = blockIdx.x & 31;          // 32 M-blocks
  const int nBlk = blockIdx.x >> 5;          // 12 N-blocks
  const int lr = lane & 15, hi = lane >> 4;

  const unsigned short* wTn   = wT + (nBlk * 64) * Cc;
  const unsigned short* arow0 = yb + (mBlk * 256 + wv * 32 + lr) * Cc;
  const unsigned short* arow1 = arow0 + 16 * Cc;

  const int scol = tid >> 2, sseg = tid & 3;
  const unsigned short* sgsrc = wTn + scol * Cc + sseg * 8;
  const unsigned lb0 = (unsigned)(size_t)(&Bsh[0][scol * BSTR + sseg * 8]);
  const unsigned lb1 = (unsigned)(size_t)(&Bsh[1][scol * BSTR + sseg * 8]);

  v8f acc[8];
#pragma unroll
  for (int j = 0; j < 8; j++)
#pragma unroll
    for (int i = 0; i < 8; i++) acc[j][i] = 0.f;

  async_copy16(lb0, sgsrc);

  for (int k0 = 0; k0 < Cc; k0 += 32) {
    const int p = (k0 >> 5) & 1;
    wait_async0();
    __syncthreads();
    if (k0 + 32 < Cc)
      async_copy16(p ? lb0 : lb1, sgsrc + k0 + 32);

    u16x16 a0 = load_a_frag(arow0 + k0, lane);
    u16x16 a1 = load_a_frag(arow1 + k0, lane);
    const unsigned short* bbuf = &Bsh[p][0];
#pragma unroll
    for (int j = 0; j < 4; j++) {
      u16x16 b = load_b_frag_lds(bbuf + (j * 16 + lr) * BSTR, lane);
      acc[j]     = wmma_bf16(a0, b, acc[j]);
      acc[4 + j] = wmma_bf16(a1, b, acc[4 + j]);
    }
  }

#pragma unroll
  for (int sub = 0; sub < 2; sub++) {
#pragma unroll
    for (int j = 0; j < 4; j++) {
      const int n    = nBlk * 64 + j * 16 + lr;
      const float bv = bias[n];
#pragma unroll
      for (int r = 0; r < 8; r++) {
        const int m = mBlk * 256 + wv * 32 + sub * 16 + r + 8 * hi;
        out[m * Cc + n] = acc[sub * 4 + j][r] + bv;
      }
    }
  }
}

// ------------------------------- launcher ----------------------------------

extern "C" void kernel_launch(void* const* d_in, const int* in_sizes, int n_in,
                              void* d_out, int out_size, void* d_ws, size_t ws_size,
                              hipStream_t stream) {
  const float* x      = (const float*)d_in[0];
  const float* w_qkv  = (const float*)d_in[1];
  const float* b_qkv  = (const float*)d_in[2];
  const float* w_proj = (const float*)d_in[3];
  const float* b_proj = (const float*)d_in[4];
  float* out = (float*)d_out;

  size_t off = 0;
  auto alloc = [&](size_t bytes) -> void* {
    void* p = (char*)d_ws + off;
    off += (bytes + 255) & ~(size_t)255;
    return p;
  };
  unsigned short* xb     = (unsigned short*)alloc((size_t)BT * Cc * 2);
  unsigned short* wqkvT  = (unsigned short*)alloc((size_t)C3 * Cc * 2);
  unsigned short* wprojT = (unsigned short*)alloc((size_t)Cc * Cc * 2);
  unsigned short* Q      = (unsigned short*)alloc((size_t)BT * Cc * 2);
  unsigned short* K      = (unsigned short*)alloc((size_t)BT * Cc * 2);
  unsigned short* V      = (unsigned short*)alloc((size_t)BT * Cc * 2);
  unsigned short* Y      = (unsigned short*)alloc((size_t)BT * Cc * 2);
  (void)ws_size; (void)in_sizes; (void)n_in; (void)out_size;

  cvt_f32_bf16<<<4096, 256, 0, stream>>>(x, xb, BT * Cc);
  transpose_f32_bf16<<<2048, 256, 0, stream>>>(w_qkv, wqkvT, Cc, C3);
  transpose_f32_bf16<<<1024, 256, 0, stream>>>(w_proj, wprojT, Cc, Cc);

  // QKV: 32 M-blocks x 36 N-blocks, 256 threads (8 waves) each
  qkv_gemm<<<32 * 36, 256, 0, stream>>>(xb, wqkvT, b_qkv, Q, K, V);

  // Attention: 48 (b,h) x 128 q-tiles = 6144 waves, 4 waves/block
  attn_fwd<<<(48 * 128) / 4, 128, 0, stream>>>(Q, K, V, Y);

  // Proj: 32 M-blocks x 12 N-blocks
  proj_gemm<<<32 * 12, 256, 0, stream>>>(Y, wprojT, b_proj, out);
}
